// DictionaryLearningWithClassifier_9414568313136
// MI455X (gfx1250) — matrix operand order repaired and checked by reading, not compiled, not run on hardware
//
#include <hip/hip_runtime.h>

#define BATCH    8192
#define FDIM     512
#define NATOMS   4096
#define SPARSITY 32

typedef __bf16 bf16_t;
typedef bf16_t v16bf __attribute__((ext_vector_type(16)));
typedef float  v8f   __attribute__((ext_vector_type(8)));

// float -> bf16 (round to nearest even), stored as raw u16
static __device__ __forceinline__ unsigned short f2bf(float f) {
    unsigned int u = __float_as_uint(f);
    unsigned int r = u + 0x7FFFu + ((u >> 16) & 1u);
    return (unsigned short)(r >> 16);
}

union FragU { uint4 u[2]; v16bf v; };

// A-fragment (16x32 bf16), CDNA5 layout:
// lane L (0..15): row = rbase + L, elems j<8 -> K = k0 + j, j>=8 -> K = k0+16+(j-8)
// lane L (16..31): same rows, K halves offset by 8 (kb = (L>>4)*8)
static __device__ __forceinline__ v16bf load_frag_a(const unsigned short* __restrict__ A,
                                                    int lda, int rbase, int k0) {
    const int lane = threadIdx.x & 31;
    const int row  = rbase + (lane & 15);
    const int kb   = (lane >> 4) * 8;
    const unsigned short* p = A + (long long)row * lda + k0 + kb;
    FragU t;
    t.u[0] = *(const uint4*)(p);        // K = k0+kb .. +7
    t.u[1] = *(const uint4*)(p + 16);   // K = k0+16+kb .. +7
    return t.v;
}

// B-fragment (32x16 bf16): lane L holds column N = cbase + (L&15),
// elems j<8 -> K = k0+kb+j, j>=8 -> K = k0+16+kb+(j-8), kb = (L>>4)*8.  Loaded from the TRANSPOSED
// dictionary DTb (rows = atoms, contiguous K) so the load is contiguous.
static __device__ __forceinline__ v16bf load_frag_b(const unsigned short* __restrict__ BT,
                                                    int ldb, int cbase, int k0) {
    const int lane = threadIdx.x & 31;
    const int col  = cbase + (lane & 15);
    const int kb   = (lane >> 4) * 8;
    const unsigned short* p = BT + (long long)col * ldb + k0 + kb;
    FragU t;
    t.u[0] = *(const uint4*)(p);
    t.u[1] = *(const uint4*)(p + 16);
    return t.v;
}

// ---------------- Kernel 1: prep (transpose + bf16 convert) ----------------
__global__ void mp_prep_kernel(const float* __restrict__ X, const float* __restrict__ D,
                               unsigned short* __restrict__ DTb, unsigned short* __restrict__ Xb,
                               float* __restrict__ DTf) {
    const long long ND = (long long)FDIM * NATOMS;   // 2M
    const long long NX = (long long)BATCH * FDIM;    // 4M
    long long t = (long long)blockIdx.x * blockDim.x + threadIdx.x;
    if (t < ND) {
        int a = (int)(t & (NATOMS - 1));   // atom (fast -> coalesced D read)
        int k = (int)(t >> 12);            // feature
        float v = D[(long long)k * NATOMS + a];
        DTf[(long long)a * FDIM + k] = v;
        DTb[(long long)a * FDIM + k] = f2bf(v);
    } else if (t < ND + NX) {
        long long i = t - ND;
        Xb[i] = f2bf(X[i]);
    }
}

// ---------------- Kernel 2: G = D^T D  (A = DTb rows, B = D = DTb^T) -------
__global__ __launch_bounds__(256) void mp_gram_kernel(const unsigned short* __restrict__ DTb,
                                                      float* __restrict__ G) {
    const int wave = threadIdx.x >> 5;
    const int lane = threadIdx.x & 31;
    const int tile = blockIdx.x * 8 + wave;       // 65536 tiles of 16x16
    const int ti = tile >> 8;                     // 256 tiles per dim
    const int tj = tile & 255;
    v8f acc = {};
    for (int k0 = 0; k0 < FDIM; k0 += 32) {
        v16bf a = load_frag_a(DTb, FDIM, ti * 16, k0);
        v16bf b = load_frag_b(DTb, FDIM, tj * 16, k0);
        acc = __builtin_amdgcn_wmma_f32_16x16x32_bf16(false, a, false, b,
                                                      (short)0, acc, false, false);
    }
    const int col   = tj * 16 + (lane & 15);
    const int rbase = ti * 16 + 8 * (lane >> 4);
#pragma unroll
    for (int v = 0; v < 8; ++v)
        G[(long long)(rbase + v) * NATOMS + col] = acc[v];
}

// ---------------- Kernel 3: fused init-proj (WMMA) + matching pursuit ------
// 512 threads = 16 wave32s; 16 samples per block; proj is LDS-resident.
__global__ __launch_bounds__(512) void mp_pursuit_kernel(const unsigned short* __restrict__ Xb,
                                                         const unsigned short* __restrict__ DTb,
                                                         const float* __restrict__ DTf,
                                                         const float* __restrict__ G,
                                                         float* __restrict__ out) {
    extern __shared__ float proj[];               // 16 * 4096 fp32 = 256 KB
    const int wave = threadIdx.x >> 5;
    const int lane = threadIdx.x & 31;
    const int m0   = blockIdx.x * 16;             // first sample of this tile

    // Phase 1: proj[16][4096] = X_tile @ D via bf16 WMMA (B frags from DTb)
    for (int i = 0; i < 16; ++i) {
        const int nt = wave + 16 * i;             // column tile 0..255
        v8f acc = {};
        for (int k0 = 0; k0 < FDIM; k0 += 32) {
            v16bf a = load_frag_a(Xb,  FDIM, m0,      k0);
            v16bf b = load_frag_b(DTb, FDIM, nt * 16, k0);
            acc = __builtin_amdgcn_wmma_f32_16x16x32_bf16(false, a, false, b,
                                                          (short)0, acc, false, false);
        }
        const int col   = nt * 16 + (lane & 15);
        const int rbase = 8 * (lane >> 4);
#pragma unroll
        for (int v = 0; v < 8; ++v)
            proj[(rbase + v) * NATOMS + col] = acc[v];
    }
    __syncthreads();

    // Phase 2: wave w runs matching pursuit on sample (m0 + w), proj in LDS,
    // Gram rows from L2-resident G (proj_{t+1} = proj_t - coef * G[best,:]).
    float* p = proj + wave * NATOMS;
    float rec[16];
#pragma unroll
    for (int k = 0; k < 16; ++k) rec[k] = 0.0f;

    for (int it = 0; it < SPARSITY; ++it) {
        // lane-strided argmax of |proj| (first index wins ties)
        float bv = -1.0f; int bi = 0;
        for (int j = lane; j < NATOMS; j += 32) {
            float v = fabsf(p[j]);
            if (v > bv) { bv = v; bi = j; }
        }
#pragma unroll
        for (int off = 16; off > 0; off >>= 1) {
            float ov = __shfl_xor(bv, off, 32);
            int   oi = __shfl_xor(bi, off, 32);
            if (ov > bv || (ov == bv && oi < bi)) { bv = ov; bi = oi; }
        }
        const float coef = p[bi];                 // pre-update projection value

        // rank-1 Gram update of projections (16 KB row, L2-resident)
        const float* __restrict__ grow = G + (long long)bi * NATOMS;
#pragma unroll 4
        for (int j = lane; j < NATOMS; j += 32)
            p[j] -= coef * grow[j];

        // accumulate reconstruction in registers: rec += coef * D[:,best]
        const float* __restrict__ drow = DTf + (long long)bi * FDIM;
#pragma unroll
        for (int k = 0; k < 16; ++k)
            rec[k] += coef * drow[k * 32 + lane];
    }

    // Phase 3: write recon
    float* orow = out + (long long)(m0 + wave) * FDIM;
#pragma unroll
    for (int k = 0; k < 16; ++k)
        orow[k * 32 + lane] = rec[k];
}

extern "C" void kernel_launch(void* const* d_in, const int* in_sizes, int n_in,
                              void* d_out, int out_size, void* d_ws, size_t ws_size,
                              hipStream_t stream) {
    const float* X = (const float*)d_in[0];            // [8192, 512]
    const float* D = (const float*)d_in[1];            // [512, 4096]
    float* out = (float*)d_out;                        // [8192, 512]

    char* ws = (char*)d_ws;
    unsigned short* DTb = (unsigned short*)(ws);                   //  4 MB bf16 D^T
    unsigned short* Xb  = (unsigned short*)(ws + (4ull  << 20));   //  8 MB bf16 X
    float*          DTf = (float*)         (ws + (12ull << 20));   //  8 MB fp32 D^T
    float*          G   = (float*)         (ws + (20ull << 20));   // 64 MB fp32 Gram

    const int ldsBytes = 16 * NATOMS * (int)sizeof(float);         // 256 KB
    (void)hipFuncSetAttribute((const void*)mp_pursuit_kernel,
                              hipFuncAttributeMaxDynamicSharedMemorySize, ldsBytes);

    const long long totalPrep = (long long)FDIM * NATOMS + (long long)BATCH * FDIM;
    const int prepBlocks = (int)((totalPrep + 255) / 256);
    mp_prep_kernel<<<prepBlocks, 256, 0, stream>>>(X, D, DTb, Xb, DTf);

    const int gramBlocks = (NATOMS / 16) * (NATOMS / 16) / 8;      // 8192
    mp_gram_kernel<<<gramBlocks, 256, 0, stream>>>(DTb, G);

    mp_pursuit_kernel<<<BATCH / 16, 512, ldsBytes, stream>>>(Xb, DTb, DTf, G, out);
}
